// CrossAttention1_63608465654185
// MI455X (gfx1250) — matrix operand order repaired and checked by reading, not compiled, not run on hardware
//
#include <hip/hip_runtime.h>
#include <hip/hip_bf16.h>

#define LDS3 __attribute__((address_space(3)))

typedef __attribute__((ext_vector_type(16))) _Float16 v16h;
typedef __attribute__((ext_vector_type(8)))  _Float16 v8h;
typedef __attribute__((ext_vector_type(4)))  _Float16 v4h;
typedef __attribute__((ext_vector_type(8)))  float    v8f;
typedef __attribute__((ext_vector_type(4)))  float    v4f;
typedef LDS3 _Float16 lh;   // f16 in LDS address space (32-bit addressing, no null checks)

// Problem constants (from reference): Bn=2048, L=8*8=64, C=256, NH=8, HD=32
constexpr int kL  = 64;
constexpr int kC  = 256;
constexpr int kHD = 32;
constexpr float kScale = 0.1767766952966369f; // 32^-0.5

// LDS row strides (in halfs), padded to break 64-bank conflicts
constexpr int XS  = kC + 8;   // 264: xh / yh / oh rows
constexpr int QSS = kHD + 8;  // 40:  qs / ks rows (64 x 32)
constexpr int VSS = kL + 8;   // 72:  vsT rows (32 x 64)
constexpr int ASS = kL + 8;   // 72:  attn rows (64 x 64)

constexpr int PERHEAD_H = kL * QSS + kL * QSS + kHD * VSS + kL * ASS; // 12032 halfs
constexpr int SMEM_HALF = 2 * kL * XS + 8 * PERHEAD_H;                // 130048 halfs
constexpr int SMEM_BYTE = SMEM_HALF * 2;                              // 260096 B

constexpr int WMAT = kC * kC;          // 65536 elements per weight matrix
constexpr size_t WS_NEED = (size_t)4 * WMAT * 2;  // f16 copies of 4 weights

// ---- WMMA fragment builders (contiguous k order, not the ISA table order; A and B match) ----

// A (16x32 f16): lane m = lane&15; halfs 0..15 -> K = 16*(lane>>4)+j
// contiguous. Two contiguous 16B LDS loads.
static __device__ inline v16h frag_a_lds(const lh* base, int stride, int lane) {
  const lh* p = base + (lane & 15) * stride + ((lane >> 4) << 4);
  union { v16h v; v8h h[2]; } u;
  u.h[0] = *(const LDS3 v8h*)(p);
  u.h[1] = *(const LDS3 v8h*)(p + 8);
  return u.v;
}

// B (32x16 f16): column N = lane&15, K = 16*(lane>>4) + 0..15 contiguous.
// Source row n of a row-major matrix supplies B column n (i.e. B = srcT).
static __device__ inline v16h frag_b_lds(const lh* base, int stride, int lane) {
  const lh* p = base + (lane & 15) * stride + ((lane >> 4) << 4);
  union { v16h v; v8h h[2]; } u;
  u.h[0] = *(const LDS3 v8h*)(p);
  u.h[1] = *(const LDS3 v8h*)(p + 8);
  return u.v;
}

// B fragment from a row-major f16 weight matrix in global memory.
static __device__ inline v16h frag_b_w16(const _Float16* __restrict__ w, int nglob,
                                         int kbase, int lane) {
  const _Float16* p = w + (size_t)nglob * kC + kbase + ((lane >> 4) << 4);
  union { v16h v; v8h h[2]; } u;
  u.h[0] = *(const v8h*)(p);
  u.h[1] = *(const v8h*)(p + 8);
  return u.v;
}

// Fallback: B fragment from a row-major f32 weight matrix (convert on load).
static __device__ inline v16h frag_b_w32(const float* __restrict__ w, int nglob,
                                         int kbase, int lane) {
  const float* p = w + (size_t)nglob * kC + kbase + ((lane >> 4) << 4);
  const v4f f0 = *(const v4f*)(p + 0);
  const v4f f1 = *(const v4f*)(p + 4);
  const v4f f2 = *(const v4f*)(p + 8);
  const v4f f3 = *(const v4f*)(p + 12);
  v16h r;
#pragma unroll
  for (int i = 0; i < 4; ++i) { r[i] = (_Float16)f0[i]; r[4+i] = (_Float16)f1[i];
                                r[8+i] = (_Float16)f2[i]; r[12+i] = (_Float16)f3[i]; }
  return r;
}

static __device__ inline v8f wmma_f16(v16h a, v16h b, v8f c) {
  return __builtin_amdgcn_wmma_f32_16x16x32_f16(false, a, false, b, (short)0, c,
                                                false, false);
}

// One head projection: dst(64x32) = src(64x256) @ W[hbase:hbase+32,:]^T + bias,
// optionally scaled, optionally stored transposed (32x64) into LDS.
template <bool WF16>
static __device__ inline void project_head(const lh* src,
                                           const float* __restrict__ wf,
                                           const _Float16* __restrict__ wh,
                                           const float* __restrict__ bias,
                                           int hbase, lh* dst, int dstride,
                                           bool transposed, float scale, int lane) {
  v8f acc[4][2];
#pragma unroll
  for (int mt = 0; mt < 4; ++mt)
#pragma unroll
    for (int nt = 0; nt < 2; ++nt) {
      const float bv = bias[hbase + nt * 16 + (lane & 15)];
#pragma unroll
      for (int i = 0; i < 8; ++i) acc[mt][nt][i] = bv;
    }

#pragma unroll
  for (int kk = 0; kk < 8; ++kk) {
    v16h a[4];
#pragma unroll
    for (int mt = 0; mt < 4; ++mt)
      a[mt] = frag_a_lds(src + mt * 16 * XS + kk * 32, XS, lane);
#pragma unroll
    for (int nt = 0; nt < 2; ++nt) {
      const int nglob = hbase + nt * 16 + (lane & 15);
      v16h bf;
      if constexpr (WF16) bf = frag_b_w16(wh, nglob, kk * 32, lane);
      else                bf = frag_b_w32(wf, nglob, kk * 32, lane);
#pragma unroll
      for (int mt = 0; mt < 4; ++mt)
        acc[mt][nt] = wmma_f16(a[mt], bf, acc[mt][nt]);
    }
  }

#pragma unroll
  for (int mt = 0; mt < 4; ++mt)
#pragma unroll
    for (int nt = 0; nt < 2; ++nt)
#pragma unroll
      for (int r = 0; r < 8; ++r) {
        const int row = mt * 16 + r + ((lane >> 4) << 3);
        const int col = nt * 16 + (lane & 15);
        const _Float16 v = (_Float16)(acc[mt][nt][r] * scale);
        if (transposed) dst[col * dstride + row] = v;
        else            dst[row * dstride + col] = v;
      }
}

// Convert the four 256x256 f32 weight matrices to f16 in workspace (once/launch).
__global__ __launch_bounds__(256) void cvt_w_kernel(
    const float* __restrict__ qw, const float* __restrict__ kw,
    const float* __restrict__ vw, const float* __restrict__ pw,
    _Float16* __restrict__ dst) {
  const int i = (blockIdx.x * 256 + threadIdx.x) * 4;
  if (i >= WMAT) return;
  const v4f a = *(const v4f*)(qw + i);
  const v4f b = *(const v4f*)(kw + i);
  const v4f c = *(const v4f*)(vw + i);
  const v4f d = *(const v4f*)(pw + i);
  v4h ha, hb, hc, hd;
#pragma unroll
  for (int j = 0; j < 4; ++j) { ha[j] = (_Float16)a[j]; hb[j] = (_Float16)b[j];
                                hc[j] = (_Float16)c[j]; hd[j] = (_Float16)d[j]; }
  *(v4h*)(dst + 0 * WMAT + i) = ha;
  *(v4h*)(dst + 1 * WMAT + i) = hb;
  *(v4h*)(dst + 2 * WMAT + i) = hc;
  *(v4h*)(dst + 3 * WMAT + i) = hd;
}

template <bool WF16>
__global__ __launch_bounds__(256) void ca_fused_kernel(
    const float* __restrict__ x, const float* __restrict__ y,
    const float* __restrict__ qw, const float* __restrict__ kw,
    const float* __restrict__ vw, const float* __restrict__ pw,
    const _Float16* __restrict__ wh,   // f16 weights base (q,k,v,p) or null
    const float* __restrict__ qb, const float* __restrict__ kb,
    const float* __restrict__ vb, const float* __restrict__ pb,
    float* __restrict__ out) {
  extern __shared__ _Float16 dsm[];
  lh* smem = (lh*)dsm;                 // AS3->generic->AS3 folds away
  lh* yh = smem;                       // 64 x XS (f16 yf)
  lh* xh = smem + kL * XS;             // 64 x XS (f16 xf); reused as oh later
  lh* headbuf = smem + 2 * kL * XS;

  const int tid  = threadIdx.x;
  const int lane = tid & 31;
  const int h    = tid >> 5;           // wave == head
  const int b    = blockIdx.x;

  const float* __restrict__ xg = x + (size_t)b * (kL * kC);
  const float* __restrict__ yg = y + (size_t)b * (kL * kC);

  // ---- Stage xf, yf to LDS as f16 (NT loads: streamed once, keep L2 for weights)
  for (int i = tid * 4; i < kL * kC; i += 256 * 4) {
    const int row = i >> 8, col = i & 255;
    const v4f vx = __builtin_nontemporal_load((const v4f*)(xg + i));
    const v4f vy = __builtin_nontemporal_load((const v4f*)(yg + i));
    lh* dx = xh + row * XS + col;
    lh* dy = yh + row * XS + col;
#pragma unroll
    for (int j = 0; j < 4; ++j) { dx[j] = (_Float16)vx[j]; dy[j] = (_Float16)vy[j]; }
  }
  __syncthreads();

  lh* qs  = headbuf + h * PERHEAD_H;
  lh* ks  = qs + kL * QSS;
  lh* vsT = ks + kL * QSS;             // stored transposed: 32 x 64
  lh* asb = vsT + kHD * VSS;           // attn probs: 64 x 64
  const int hbase = h * kHD;

  // ---- Q/K/V head projections (each wave does its own head) ----
  project_head<WF16>(yh, qw, wh + 0 * WMAT, qb, hbase, qs,  QSS, false, kScale, lane);
  project_head<WF16>(xh, kw, wh + 1 * WMAT, kb, hbase, ks,  QSS, false, 1.0f,  lane);
  project_head<WF16>(xh, vw, wh + 2 * WMAT, vb, hbase, vsT, VSS, true,  1.0f,  lane);
  __syncthreads();  // all waves done reading xh before it becomes oh

  // ---- attn = q @ k^T (64x64, K=32: one WMMA per tile) ----
  v8f att[4][4];
#pragma unroll
  for (int mt = 0; mt < 4; ++mt) {
    const v16h a = frag_a_lds(qs + mt * 16 * QSS, QSS, lane);
#pragma unroll
    for (int nt = 0; nt < 4; ++nt) {
      const v16h bf = frag_b_lds(ks + nt * 16 * QSS, QSS, lane);
      v8f cz;
#pragma unroll
      for (int i = 0; i < 8; ++i) cz[i] = 0.0f;
      att[mt][nt] = wmma_f16(a, bf, cz);
    }
  }

  // ---- softmax_plus_one per attn row; store probs f16 to LDS ----
  // Row M = 16*mt + r + 8*(lane>>4) lives in one 16-lane half-wave.
#pragma unroll
  for (int mt = 0; mt < 4; ++mt)
#pragma unroll
    for (int r = 0; r < 8; ++r) {
      const float v0 = att[mt][0][r], v1 = att[mt][1][r];
      const float v2 = att[mt][2][r], v3 = att[mt][3][r];
      float m = fmaxf(fmaxf(v0, v1), fmaxf(v2, v3));
#pragma unroll
      for (int s = 8; s >= 1; s >>= 1) m = fmaxf(m, __shfl_xor(m, s, 32));
      const float e0 = __expf(v0 - m), e1 = __expf(v1 - m);
      const float e2 = __expf(v2 - m), e3 = __expf(v3 - m);
      float sum = e0 + e1 + e2 + e3;
#pragma unroll
      for (int s = 8; s >= 1; s >>= 1) sum += __shfl_xor(sum, s, 32);
      const float inv = 1.0f / (sum + 1.0f);   // softmax_plus_one
      const int row = mt * 16 + r + ((lane >> 4) << 3);
      const int cb  = lane & 15;
      asb[row * ASS +  0 + cb] = (_Float16)(e0 * inv);
      asb[row * ASS + 16 + cb] = (_Float16)(e1 * inv);
      asb[row * ASS + 32 + cb] = (_Float16)(e2 * inv);
      asb[row * ASS + 48 + cb] = (_Float16)(e3 * inv);
    }

  // ---- y1 = attn @ v (64x32, K=64: 2 K-steps); residual; concat heads ----
  v8f yacc[4][2];
#pragma unroll
  for (int mt = 0; mt < 4; ++mt)
#pragma unroll
    for (int nt = 0; nt < 2; ++nt)
#pragma unroll
      for (int i = 0; i < 8; ++i) yacc[mt][nt][i] = 0.0f;

#pragma unroll
  for (int kk = 0; kk < 2; ++kk) {
    v16h a[4];
#pragma unroll
    for (int mt = 0; mt < 4; ++mt)
      a[mt] = frag_a_lds(asb + mt * 16 * ASS + kk * 32, ASS, lane);
#pragma unroll
    for (int nt = 0; nt < 2; ++nt) {
      const v16h bf = frag_b_lds(vsT + nt * 16 * VSS + kk * 32, VSS, lane);
#pragma unroll
      for (int mt = 0; mt < 4; ++mt)
        yacc[mt][nt] = wmma_f16(a[mt], bf, yacc[mt][nt]);
    }
  }

  lh* oh = xh;  // overlay: xh no longer needed
#pragma unroll
  for (int mt = 0; mt < 4; ++mt)
#pragma unroll
    for (int nt = 0; nt < 2; ++nt)
#pragma unroll
      for (int r = 0; r < 8; ++r) {
        const int row = mt * 16 + r + ((lane >> 4) << 3);
        const int col = hbase + nt * 16 + (lane & 15);
        const float v = yacc[mt][nt][r] + yg[row * kC + col];  // residual (f32)
        oh[row * XS + col] = (_Float16)v;
      }
  __syncthreads();

  // ---- out = oh @ proj_w^T + proj_b ; wave h covers N-tiles {2h, 2h+1} ----
  v8f oacc[4][2];
#pragma unroll
  for (int mt = 0; mt < 4; ++mt)
#pragma unroll
    for (int nt = 0; nt < 2; ++nt) {
      const float bv = pb[(2 * h + nt) * 16 + (lane & 15)];
#pragma unroll
      for (int i = 0; i < 8; ++i) oacc[mt][nt][i] = bv;
    }

#pragma unroll
  for (int kk = 0; kk < 8; ++kk) {
    v16h a[4];
#pragma unroll
    for (int mt = 0; mt < 4; ++mt)
      a[mt] = frag_a_lds(oh + mt * 16 * XS + kk * 32, XS, lane);
#pragma unroll
    for (int nt = 0; nt < 2; ++nt) {
      const int nglob = (2 * h + nt) * 16 + (lane & 15);
      v16h bf;
      if constexpr (WF16) bf = frag_b_w16(wh + 3 * WMAT, nglob, kk * 32, lane);
      else                bf = frag_b_w32(pw, nglob, kk * 32, lane);
#pragma unroll
      for (int mt = 0; mt < 4; ++mt)
        oacc[mt][nt] = wmma_f16(a[mt], bf, oacc[mt][nt]);
    }
  }

  float* __restrict__ og = out + (size_t)b * (kL * kC);
#pragma unroll
  for (int mt = 0; mt < 4; ++mt)
#pragma unroll
    for (int nt = 0; nt < 2; ++nt)
#pragma unroll
      for (int r = 0; r < 8; ++r) {
        const int row = mt * 16 + r + ((lane >> 4) << 3);
        const int col = (2 * h + nt) * 16 + (lane & 15);
        __builtin_nontemporal_store(oacc[mt][nt][r], og + row * kC + col);
      }
}

extern "C" void kernel_launch(void* const* d_in, const int* in_sizes, int n_in,
                              void* d_out, int out_size, void* d_ws, size_t ws_size,
                              hipStream_t stream) {
  const float* x  = (const float*)d_in[0];
  const float* y  = (const float*)d_in[1];
  // d_in[2] (z) is unused by the reference forward
  const float* qw = (const float*)d_in[3];
  const float* qb = (const float*)d_in[4];
  const float* kw = (const float*)d_in[5];
  const float* kb = (const float*)d_in[6];
  const float* vw = (const float*)d_in[7];
  const float* vb = (const float*)d_in[8];
  const float* pw = (const float*)d_in[9];
  const float* pb = (const float*)d_in[10];
  float* out = (float*)d_out;

  const int Bn = in_sizes[0] / (kL * kC);  // 2048

  if (ws_size >= WS_NEED) {
    _Float16* wh = (_Float16*)d_ws;
    cvt_w_kernel<<<WMAT / (256 * 4), 256, 0, stream>>>(qw, kw, vw, pw, wh);
    hipFuncSetAttribute((const void*)&ca_fused_kernel<true>,
                        hipFuncAttributeMaxDynamicSharedMemorySize, SMEM_BYTE);
    ca_fused_kernel<true><<<Bn, 256, SMEM_BYTE, stream>>>(
        x, y, qw, kw, vw, pw, wh, qb, kb, vb, pb, out);
  } else {
    hipFuncSetAttribute((const void*)&ca_fused_kernel<false>,
                        hipFuncAttributeMaxDynamicSharedMemorySize, SMEM_BYTE);
    ca_fused_kernel<false><<<Bn, 256, SMEM_BYTE, stream>>>(
        x, y, qw, kw, vw, pw, (const _Float16*)nullptr, qb, kb, vb, pb, out);
  }
}
